// CausalSelfAttention_9320079032991
// MI455X (gfx1250) — matrix operand order repaired and checked by reading, not compiled, not run on hardware
//
#include <hip/hip_runtime.h>

// ---------------------------------------------------------------------------
// Causal self-attention for MI455X (gfx1250, wave32, WMMA bf16 path)
// B=4, S=2048, D=1024, H=16, HD=64
//
// Pipeline:
//   0) convert_kernel:  x, Wq, Wk, Wv, Wo  -> bf16 copies in workspace
//   1) qkv_gemm_kernel: Q/K/V = xb @ W^T + b  (async-LDS A panel, 16x64/wave)
//   2) attn_kernel:     flash attention, online softmax, LDS P-transpose
//   3) out_gemm_kernel: out = attn @ Wo^T + bo (fp32 result)
// ---------------------------------------------------------------------------

typedef __attribute__((ext_vector_type(16))) __bf16 v16bf;
typedef __attribute__((ext_vector_type(8)))  float  v8f;

#define WMMA_BF16(a, b, c) \
  __builtin_amdgcn_wmma_f32_16x16x32_bf16(false, (a), false, (b), (short)0, (c), false, false)

constexpr int Bc  = 4;
constexpr int Sc  = 2048;
constexpr int Dc  = 1024;
constexpr int Hc  = 16;
constexpr int HDc = 64;
constexpr int Mtot = Bc * Sc;  // 8192 rows
constexpr float ATTN_SCALE = 0.125f;  // 1/sqrt(64)

// Load a 16-element bf16 fragment as two 16-byte chunks.
static __device__ __forceinline__ v16bf ld_frag(const __bf16* p0, const __bf16* p1) {
  v16bf r;
  reinterpret_cast<uint4*>(&r)[0] = *reinterpret_cast<const uint4*>(p0);
  reinterpret_cast<uint4*>(&r)[1] = *reinterpret_cast<const uint4*>(p1);
  return r;
}

// Async global->LDS copy of 16 bytes (CDNA5 GLOBAL_LOAD_ASYNC_TO_LDS_B128,
// GVS mode: saddr = 64-bit SGPR base, vaddr = 32-bit byte offset,
// vdst = LDS byte address). Tracked by ASYNCcnt.
static __device__ __forceinline__ void async_copy_b128(uint32_t lds_byte,
                                                       const void* gbase,
                                                       uint32_t gbyte_off) {
  asm volatile("global_load_async_to_lds_b128 %0, %1, %2"
               :: "v"(lds_byte), "v"(gbyte_off), "s"(gbase)
               : "memory");
}

static __device__ __forceinline__ void wait_async_all() {
  asm volatile("s_wait_asynccnt 0x0" ::: "memory");
}

// ---------------------------------------------------------------------------
// Kernel 0: elementwise f32 -> bf16 (8 elements / thread, b128 in, b128 out)
// ---------------------------------------------------------------------------
__global__ __launch_bounds__(256) void convert_kernel(const float* __restrict__ src,
                                                      __bf16* __restrict__ dst,
                                                      int n8) {
  const int i = blockIdx.x * 256 + threadIdx.x;
  if (i >= n8) return;
  const float4 u = reinterpret_cast<const float4*>(src)[2 * i];
  const float4 v = reinterpret_cast<const float4*>(src)[2 * i + 1];
  __bf16 o[8];
  o[0] = (__bf16)u.x; o[1] = (__bf16)u.y; o[2] = (__bf16)u.z; o[3] = (__bf16)u.w;
  o[4] = (__bf16)v.x; o[5] = (__bf16)v.y; o[6] = (__bf16)v.z; o[7] = (__bf16)v.w;
  reinterpret_cast<uint4*>(dst)[i] = *reinterpret_cast<const uint4*>(o);
}

// ---------------------------------------------------------------------------
// Shared GEMM core: one wave computes a 16(M) x 64(N) tile, K = 1024.
// A panel (16 x 1024 bf16, 32KB) lives in LDS; B rows read bf16 from global.
// 4 WMMAs per K-step; A fragment reused 4x.
// ---------------------------------------------------------------------------
static __device__ __forceinline__ void gemm_16x64(const __bf16* ldsA,
                                                  const __bf16* __restrict__ Wb,
                                                  int nbase, int lane, v8f c[4]) {
  const int lm = lane & 15, half = lane >> 4, klo = half * 8;
  const __bf16* br[4];
#pragma unroll
  for (int g = 0; g < 4; ++g)
    br[g] = Wb + (size_t)(nbase + g * 16 + lm) * Dc + half * 8;

  const __bf16* arow = ldsA + lm * Dc;
  for (int k0 = 0; k0 < Dc; k0 += 32) {
    const v16bf a = ld_frag(arow + k0 + klo, arow + k0 + 16 + klo);  // ds_load
#pragma unroll
    for (int g = 0; g < 4; ++g) {
      __builtin_prefetch(br[g] + k0 + 256, 0, 0);
      const v16bf b = ld_frag(br[g] + k0, br[g] + k0 + 16);
      c[g] = WMMA_BF16(a, b, c[g]);
    }
  }
}

// Stage 16 contiguous rows (16 x 1024 bf16 = 32KB, contiguous in memory)
// into LDS with async copies: 256 threads x 8 chunks of 16B.
static __device__ __forceinline__ void stage_A_panel(__bf16* ldsA,
                                                     const __bf16* gsrc) {
  const uint32_t lds_base = (uint32_t)(size_t)ldsA;  // low 32 bits = LDS offset
  const uint32_t tid = threadIdx.x;
#pragma unroll
  for (int i = 0; i < 8; ++i) {
    const uint32_t chunk = i * 256 + tid;      // 2048 chunks of 16B
    const uint32_t off = chunk * 16;
    async_copy_b128(lds_base + off, (const void*)gsrc, off);
  }
  wait_async_all();
  __syncthreads();
}

// ---------------------------------------------------------------------------
// Kernel 1: fused Q/K/V projection (bf16 in, bf16 out, f32 accum).
// Q: [B,H,S,HD] pre-scaled by 1/sqrt(HD); K: [B,H,S,HD]; V: [B,H,HD,S].
// Grid: 3 proj x 512 mtiles x 2 n-halves. Block = 8 waves x 64 N = 512 N.
// ---------------------------------------------------------------------------
__global__ __launch_bounds__(256) void qkv_gemm_kernel(
    const __bf16* __restrict__ xb,
    const __bf16* __restrict__ Wqb, const float* __restrict__ bq,
    const __bf16* __restrict__ Wkb, const float* __restrict__ bk,
    const __bf16* __restrict__ Wvb, const float* __restrict__ bv,
    __bf16* __restrict__ Qb, __bf16* __restrict__ Kb, __bf16* __restrict__ Vt) {
  __shared__ __bf16 As[16 * Dc];  // 32KB A panel

  const int lane   = threadIdx.x & 31;
  const int waveId = threadIdx.x >> 5;
  const int bx   = blockIdx.x;
  const int proj = bx / ((Mtot / 16) * 2);
  const int rem  = bx % ((Mtot / 16) * 2);
  const int mt   = rem >> 1;
  const int nh   = rem & 1;
  const int nbase = nh * 512 + waveId * 64;

  const __bf16* W; const float* bias; float scale;
  if (proj == 0)      { W = Wqb; bias = bq; scale = ATTN_SCALE; }
  else if (proj == 1) { W = Wkb; bias = bk; scale = 1.0f; }
  else                { W = Wvb; bias = bv; scale = 1.0f; }

  stage_A_panel(As, xb + (size_t)mt * 16 * Dc);

  v8f c[4];
#pragma unroll
  for (int g = 0; g < 4; ++g)
#pragma unroll
    for (int r = 0; r < 8; ++r) c[g][r] = 0.f;

  gemm_16x64(As, W, nbase, lane, c);

  const int lm = lane & 15, half = lane >> 4;
  const int b_ = (mt * 16) / Sc;
  const int s0 = (mt * 16) % Sc;
#pragma unroll
  for (int g = 0; g < 4; ++g) {
    const int n = nbase + g * 16 + lm;
    const float bn = bias[n];
    const int h = n >> 6, hd = n & 63;
    const size_t bh = (size_t)(b_ * Hc + h);
#pragma unroll
    for (int r = 0; r < 8; ++r) {
      const float y = (c[g][r] + bn) * scale;
      const int s = s0 + half * 8 + r;
      if (proj == 0)      Qb[(bh * Sc + s) * HDc + hd] = (__bf16)y;
      else if (proj == 1) Kb[(bh * Sc + s) * HDc + hd] = (__bf16)y;
      else                Vt[(bh * HDc + hd) * Sc + s] = (__bf16)y;
    }
  }
}

// ---------------------------------------------------------------------------
// Kernel 2: flash attention. One wave owns a 16-row Q tile of one (b,h).
// ---------------------------------------------------------------------------
__global__ __launch_bounds__(256) void attn_kernel(
    const __bf16* __restrict__ Qb, const __bf16* __restrict__ Kb,
    const __bf16* __restrict__ Vt, __bf16* __restrict__ Ab) {
  __shared__ __bf16 plds[8 * 16 * 32];  // 1KB per-wave P staging tile

  const int lane   = threadIdx.x & 31;
  const int waveId = threadIdx.x >> 5;
  const int wave   = blockIdx.x * 8 + waveId;
  const int qt = wave & (Sc / 16 - 1);
  const int bh = wave >> 7;

  const int lm = lane & 15, half = lane >> 4, klo = half * 8;

  const __bf16* Qh = Qb + (size_t)bh * Sc * HDc;
  const __bf16* Kh = Kb + (size_t)bh * Sc * HDc;
  const __bf16* Vh = Vt + (size_t)bh * HDc * Sc;
  __bf16* pl = plds + waveId * (16 * 32);

  const __bf16* qrow = Qh + (size_t)(qt * 16 + lm) * HDc;
  const v16bf aq0 = ld_frag(qrow + klo,      qrow + 16 + klo);
  const v16bf aq1 = ld_frag(qrow + 32 + klo, qrow + 48 + klo);

  v8f acc[4];
  float mrow[8], lrow[8];
#pragma unroll
  for (int ci = 0; ci < 4; ++ci)
#pragma unroll
    for (int r = 0; r < 8; ++r) acc[ci][r] = 0.f;
#pragma unroll
  for (int r = 0; r < 8; ++r) { mrow[r] = -1e30f; lrow[r] = 0.f; }

  const int nkb = (qt * 16 + 16 + 31) >> 5;
  for (int kb = 0; kb < nkb; ++kb) {
    const int k0 = kb * 32;

    v8f sc[2];
#pragma unroll
    for (int t = 0; t < 2; ++t) {
      const __bf16* krow = Kh + (size_t)(k0 + t * 16 + lm) * HDc + half * 8;
      const v16bf bk0 = ld_frag(krow,      krow + 16);
      const v16bf bk1 = ld_frag(krow + 32, krow + 48);
      v8f z = {0.f, 0.f, 0.f, 0.f, 0.f, 0.f, 0.f, 0.f};
      z = WMMA_BF16(aq0, bk0, z);
      z = WMMA_BF16(aq1, bk1, z);
      sc[t] = z;
    }

    const int j0 = k0 + lm;
#pragma unroll
    for (int r = 0; r < 8; ++r) {
      const int rowg = qt * 16 + half * 8 + r;
      float t0 = (j0      <= rowg) ? sc[0][r] : -1e30f;
      float t1 = (j0 + 16 <= rowg) ? sc[1][r] : -1e30f;
      float mx = fmaxf(t0, t1);
      mx = fmaxf(mx, __shfl_xor(mx, 1, 32));
      mx = fmaxf(mx, __shfl_xor(mx, 2, 32));
      mx = fmaxf(mx, __shfl_xor(mx, 4, 32));
      mx = fmaxf(mx, __shfl_xor(mx, 8, 32));
      const float mnew  = fmaxf(mrow[r], mx);
      const float alpha = __expf(mrow[r] - mnew);
      const float e0 = __expf(t0 - mnew);
      const float e1 = __expf(t1 - mnew);
      float sum = e0 + e1;
      sum += __shfl_xor(sum, 1, 32);
      sum += __shfl_xor(sum, 2, 32);
      sum += __shfl_xor(sum, 4, 32);
      sum += __shfl_xor(sum, 8, 32);
      lrow[r] = lrow[r] * alpha + sum;
      mrow[r] = mnew;
      acc[0][r] *= alpha; acc[1][r] *= alpha;
      acc[2][r] *= alpha; acc[3][r] *= alpha;
      const int row = half * 8 + r;
      pl[row * 32 + lm]      = (__bf16)e0;
      pl[row * 32 + 16 + lm] = (__bf16)e1;
    }

    asm volatile("s_wait_dscnt 0" ::: "memory");

    const v16bf ap = ld_frag(pl + lm * 32 + klo, pl + lm * 32 + 16 + klo);

#pragma unroll
    for (int ci = 0; ci < 4; ++ci) {
      const __bf16* vrow = Vh + (size_t)(ci * 16 + lm) * Sc + k0 + half * 8;
      const v16bf bv = ld_frag(vrow, vrow + 16);
      acc[ci] = WMMA_BF16(ap, bv, acc[ci]);
    }
  }

  const int b_ = bh >> 4, h = bh & 15;
#pragma unroll
  for (int r = 0; r < 8; ++r) {
    const float inv = 1.0f / lrow[r];
    const int s = qt * 16 + half * 8 + r;
#pragma unroll
    for (int ci = 0; ci < 4; ++ci) {
      const int d = h * 64 + ci * 16 + lm;
      Ab[((size_t)(b_ * Sc + s)) * Dc + d] = (__bf16)(acc[ci][r] * inv);
    }
  }
}

// ---------------------------------------------------------------------------
// Kernel 3: output projection, fp32 result.
// Grid: 512 mtiles x 2 n-halves; block = 8 waves x 64 N.
// ---------------------------------------------------------------------------
__global__ __launch_bounds__(256) void out_gemm_kernel(
    const __bf16* __restrict__ Ab, const __bf16* __restrict__ Wob,
    const float* __restrict__ bo, float* __restrict__ out) {
  __shared__ __bf16 As[16 * Dc];

  const int lane   = threadIdx.x & 31;
  const int waveId = threadIdx.x >> 5;
  const int mt = blockIdx.x >> 1;
  const int nh = blockIdx.x & 1;
  const int nbase = nh * 512 + waveId * 64;

  stage_A_panel(As, Ab + (size_t)mt * 16 * Dc);

  v8f c[4];
#pragma unroll
  for (int g = 0; g < 4; ++g)
#pragma unroll
    for (int r = 0; r < 8; ++r) c[g][r] = 0.f;

  gemm_16x64(As, Wob, nbase, lane, c);

  const int lm = lane & 15, half = lane >> 4;
#pragma unroll
  for (int g = 0; g < 4; ++g) {
    const int n = nbase + g * 16 + lm;
    const float bn = bo[n];
#pragma unroll
    for (int r = 0; r < 8; ++r) {
      const int m = mt * 16 + half * 8 + r;
      out[(size_t)m * Dc + n] = c[g][r] + bn;
    }
  }
}

// ---------------------------------------------------------------------------
extern "C" void kernel_launch(void* const* d_in, const int* in_sizes, int n_in,
                              void* d_out, int out_size, void* d_ws, size_t ws_size,
                              hipStream_t stream) {
  const float* x  = (const float*)d_in[0];
  // d_in[1] = causal mask (bool) — causality baked into attn_kernel
  const float* Wq = (const float*)d_in[2];
  const float* bq = (const float*)d_in[3];
  const float* Wk = (const float*)d_in[4];
  const float* bk = (const float*)d_in[5];
  const float* Wv = (const float*)d_in[6];
  const float* bv = (const float*)d_in[7];
  const float* Wo = (const float*)d_in[8];
  const float* bo = (const float*)d_in[9];
  float* out = (float*)d_out;

  // Workspace layout (bf16 elements)
  const size_t nX = (size_t)Mtot * Dc;       // 8M
  const size_t nW = (size_t)Dc * Dc;         // 1M
  const size_t nH = (size_t)Bc * Hc * Sc * HDc;  // 8M
  __bf16* xb  = (__bf16*)d_ws;
  __bf16* Wqb = xb + nX;
  __bf16* Wkb = Wqb + nW;
  __bf16* Wvb = Wkb + nW;
  __bf16* Wob = Wvb + nW;
  __bf16* Qb  = Wob + nW;
  __bf16* Kb  = Qb + nH;
  __bf16* Vt  = Kb + nH;
  __bf16* Ab  = Vt + nH;

  // 0) f32 -> bf16 conversions (once per launch)
  convert_kernel<<<(int)(nX / 8 / 256), 256, 0, stream>>>(x, xb, (int)(nX / 8));
  convert_kernel<<<(int)(nW / 8 / 256), 256, 0, stream>>>(Wq, Wqb, (int)(nW / 8));
  convert_kernel<<<(int)(nW / 8 / 256), 256, 0, stream>>>(Wk, Wkb, (int)(nW / 8));
  convert_kernel<<<(int)(nW / 8 / 256), 256, 0, stream>>>(Wv, Wvb, (int)(nW / 8));
  convert_kernel<<<(int)(nW / 8 / 256), 256, 0, stream>>>(Wo, Wob, (int)(nW / 8));

  // 1) QKV projections: 3 x 512 x 2 blocks
  qkv_gemm_kernel<<<3 * (Mtot / 16) * 2, 256, 0, stream>>>(
      xb, Wqb, bq, Wkb, bk, Wvb, bv, Qb, Kb, Vt);

  // 2) attention: B*H*(S/16) waves / 8 per block
  attn_kernel<<<Bc * Hc * (Sc / 16) / 8, 256, 0, stream>>>(Qb, Kb, Vt, Ab);

  // 3) output projection: 512 x 2 blocks
  out_gemm_kernel<<<(Mtot / 16) * 2, 256, 0, stream>>>(Ab, Wob, bo, out);
}
